// GINConvNet_73014444032011
// MI455X (gfx1250) — matrix operand order repaired and checked by reading, not compiled, not run on hardware
//
#include <hip/hip_runtime.h>
#include <hip/hip_bf16.h>

// ---------------- problem constants ----------------
#define N_ATOMS 100000
#define E_EDGES 400000
#define B_G     1024
#define F_IN    78
#define F_PAD   96            // F_IN padded to multiple of 32 (branch-free K loop)
#define DIM     32
#define OUTD    128
#define EMBD    128
#define VOCAB   26
#define NF      32
#define SEQ     1000
#define KCONV   8
#define PCONV   993           // SEQ - KCONV + 1
#define FLATD   31776         // NF * PCONV (= 993 WMMA K-steps, exact)
#define COMB    2304
#define LMD     1024

typedef __attribute__((ext_vector_type(16))) __bf16 v16bf;
typedef __attribute__((ext_vector_type(8)))  __bf16 v8bf;
typedef __attribute__((ext_vector_type(8)))  float  v8f;
typedef __attribute__((ext_vector_type(4)))  float  v4f;

__device__ __forceinline__ v8f wmma_bf16(v16bf a, v16bf b, v8f c) {
  return __builtin_amdgcn_wmma_f32_16x16x32_bf16(false, a, false, b, (short)0, c,
                                                 false, false);
}
__device__ __forceinline__ v16bf cat16(v8bf lo, v8bf hi) {
  return __builtin_shufflevector(lo, hi, 0, 1, 2, 3, 4, 5, 6, 7, 8, 9, 10, 11,
                                 12, 13, 14, 15);
}
__device__ __forceinline__ v8f vzero8() {
  return (v8f){0.f, 0.f, 0.f, 0.f, 0.f, 0.f, 0.f, 0.f};
}

// =====================================================================
// Generic GEMM: out = relu(A[MxK]_bf16 * Bt[NxK]_bf16^T + bias).
// One 16x(16*NT) tile per wave: NT accumulators share each A fetch,
// cutting A traffic by NT and raising wmma:vmem density.
// K must be a multiple of 32 (all call sites are).
// A operand: two contiguous 16B runs per lane (ISA 7.12.2 16-bit A layout).
// B operand: weights pre-transposed -> two 16B runs per lane, same k order as A.
// =====================================================================
template <int NT, bool OUT_BF>
__global__ __launch_bounds__(256) void gemm_bf16_wmma(
    const __bf16* __restrict__ A, const __bf16* __restrict__ Bt,
    const float* __restrict__ bias, void* __restrict__ outp, int M, int Nn,
    int Kk, int lda, int ldc, int col0) {
  const int groups_n = Nn / (16 * NT);
  const int total = (M >> 4) * groups_n;
  const int wid = blockIdx.x * (blockDim.x >> 5) + (threadIdx.x >> 5);
  if (wid >= total) return;
  const int tm = wid / groups_n, tg = wid % groups_n;
  const int lane = threadIdx.x & 31;
  const int half = lane >> 4, l16 = lane & 15;

  const __bf16* arow = A + (size_t)(tm * 16 + l16) * lda + 8 * half;
  const __bf16* brow[NT];
#pragma unroll
  for (int t = 0; t < NT; ++t)
    brow[t] = Bt + (size_t)((tg * NT + t) * 16 + l16) * Kk + 8 * half;

  v8f acc[NT];
#pragma unroll
  for (int t = 0; t < NT; ++t) acc[t] = vzero8();

  const int ksteps = Kk >> 5;
  for (int kb = 0; kb < ksteps; ++kb) {
    const int k0 = kb << 5;
    __builtin_prefetch(arow + k0 + 512, 0, 1);  // global_prefetch_b8
    v16bf a =
        cat16(*(const v8bf*)(arow + k0), *(const v8bf*)(arow + k0 + 16));
#pragma unroll
    for (int t = 0; t < NT; ++t) {
      __builtin_prefetch(brow[t] + k0 + 512, 0, 1);
      v16bf b = cat16(*(const v8bf*)(brow[t] + k0),
                      *(const v8bf*)(brow[t] + k0 + 16));
      acc[t] = wmma_bf16(a, b, acc[t]);
    }
  }
#pragma unroll
  for (int t = 0; t < NT; ++t) {
    const int nb = (tg * NT + t) * 16 + l16;
    const float bv = bias[nb];
    const int n = col0 + nb;
#pragma unroll
    for (int r = 0; r < 8; ++r) {
      const int m = tm * 16 + r + 8 * half;  // C/D layout row
      const float v = fmaxf(acc[t][r] + bv, 0.f);
      if (OUT_BF) ((__bf16*)outp)[(size_t)m * ldc + n] = (__bf16)v;
      else        ((float*)outp)[(size_t)m * ldc + n] = v;
    }
  }
}

// =====================================================================
// GIN MLP: z = relu( relu((h+agg)@Wa+ba) @ Wb + bb ), one 16-node tile
// per wave, branch-free (inputs/weights pre-padded to K multiple of 32).
// Wa_t/Wb_t are bf16, transposed [32 x Kpad]. C->A re-layout goes through
// per-wave LDS (vector ds loads) with an explicit s_wait_dscnt fence.
// =====================================================================
__global__ __launch_bounds__(256) void gin_mlp_wmma(
    const float* __restrict__ h, const float* __restrict__ agg, int inF,
    const __bf16* __restrict__ wa_t, const float* __restrict__ ba,
    const __bf16* __restrict__ wb_t, const float* __restrict__ bb,
    float* __restrict__ ztmp, int ntiles) {
  __shared__ float ylds[8][16 * 32];
  const int wslot = threadIdx.x >> 5;
  const int wid = blockIdx.x * 8 + wslot;
  if (wid >= ntiles) return;
  const int lane = threadIdx.x & 31;
  const int half = lane >> 4, l16 = lane & 15;

  const float* hr = h + (size_t)(wid * 16 + l16) * inF + 8 * half;
  const float* gr = agg + (size_t)(wid * 16 + l16) * inF + 8 * half;
  const __bf16* w0 = wa_t + (size_t)l16 * inF + 8 * half;         // cols 0-15
  const __bf16* w1 = wa_t + (size_t)(16 + l16) * inF + 8 * half;  // cols 16-31

  v8f acc0 = vzero8(), acc1 = vzero8();
  const int ksteps = inF >> 5;
  for (int kb = 0; kb < ksteps; ++kb) {
    const int k0 = kb << 5;
    v4f h0 = *(const v4f*)(hr + k0),      h1 = *(const v4f*)(hr + k0 + 4);
    v4f h2 = *(const v4f*)(hr + k0 + 16), h3 = *(const v4f*)(hr + k0 + 20);
    v4f g0 = *(const v4f*)(gr + k0),      g1 = *(const v4f*)(gr + k0 + 4);
    v4f g2 = *(const v4f*)(gr + k0 + 16), g3 = *(const v4f*)(gr + k0 + 20);
    v16bf a;
#pragma unroll
    for (int i = 0; i < 4; ++i) {
      a[i]      = (__bf16)(h0[i] + g0[i]);
      a[4 + i]  = (__bf16)(h1[i] + g1[i]);
      a[8 + i]  = (__bf16)(h2[i] + g2[i]);
      a[12 + i] = (__bf16)(h3[i] + g3[i]);
    }
    v16bf b0 = cat16(*(const v8bf*)(w0 + k0), *(const v8bf*)(w0 + k0 + 16));
    v16bf b1 = cat16(*(const v8bf*)(w1 + k0), *(const v8bf*)(w1 + k0 + 16));
    acc0 = wmma_bf16(a, b0, acc0);
    acc1 = wmma_bf16(a, b1, acc1);
  }
  // bias + relu -> stage Y[16x32] row-major in this wave's LDS slab
  float* Y = ylds[wslot];
#pragma unroll
  for (int r = 0; r < 8; ++r) {
    const int mm = r + 8 * half;
    Y[mm * 32 + l16]      = fmaxf(acc0[r] + ba[l16], 0.f);
    Y[mm * 32 + 16 + l16] = fmaxf(acc1[r] + ba[16 + l16], 0.f);
  }
  asm volatile("s_wait_dscnt 0" ::: "memory");  // wave-local LDS RAW fence

  // --- GEMM2: Y[16x32] @ Wb[32x32] (Wb_t transposed bf16) ---
  const float* yr = Y + l16 * 32 + 8 * half;
  v4f y0 = *(const v4f*)(yr),      y1 = *(const v4f*)(yr + 4);
  v4f y2 = *(const v4f*)(yr + 16), y3 = *(const v4f*)(yr + 20);
  v16bf a2;
#pragma unroll
  for (int i = 0; i < 4; ++i) {
    a2[i]      = (__bf16)y0[i];
    a2[4 + i]  = (__bf16)y1[i];
    a2[8 + i]  = (__bf16)y2[i];
    a2[12 + i] = (__bf16)y3[i];
  }
  const __bf16* u0 = wb_t + l16 * 32 + 8 * half;
  const __bf16* u1 = wb_t + (16 + l16) * 32 + 8 * half;
  v16bf c0 = cat16(*(const v8bf*)(u0), *(const v8bf*)(u0 + 16));
  v16bf c1 = cat16(*(const v8bf*)(u1), *(const v8bf*)(u1 + 16));
  v8f z0 = vzero8(), z1 = vzero8();
  z0 = wmma_bf16(a2, c0, z0);
  z1 = wmma_bf16(a2, c1, z1);
#pragma unroll
  for (int r = 0; r < 8; ++r) {
    const int m = wid * 16 + r + 8 * half;
    ztmp[(size_t)m * DIM + l16]      = fmaxf(z0[r] + bb[l16], 0.f);
    ztmp[(size_t)m * DIM + 16 + l16] = fmaxf(z1[r] + bb[16 + l16], 0.f);
  }
}

// ---------------- graph aggregation / BN / pool ----------------
__global__ void scatter_kernel(const float* __restrict__ h,
                               const int* __restrict__ ei,
                               float* __restrict__ agg, int F, int hs, int as) {
  const long long idx = (long long)blockIdx.x * blockDim.x + threadIdx.x;
  if (idx >= (long long)E_EDGES * F) return;
  const int e = (int)(idx / F), c = (int)(idx % F);
  const int s = ei[e], d = ei[E_EDGES + e];
  atomicAdd(&agg[(size_t)d * as + c], h[(size_t)s * hs + c]);
}

__global__ void bn_stats_kernel(const float* __restrict__ z,
                                float* __restrict__ stats) {  // stats[64]
  __shared__ float red[2][256];
  const int c = threadIdx.x & 31, rt = threadIdx.x >> 5;
  float s1 = 0.f, s2 = 0.f;
  for (int row = blockIdx.x * 8 + rt; row < N_ATOMS; row += gridDim.x * 8) {
    const float v = z[(size_t)row * 32 + c];
    s1 += v; s2 += v * v;
  }
  red[0][threadIdx.x] = s1;
  red[1][threadIdx.x] = s2;
  __syncthreads();
  if (threadIdx.x < 32) {
    float t1 = 0.f, t2 = 0.f;
    for (int r = 0; r < 8; ++r) {
      t1 += red[0][r * 32 + threadIdx.x];
      t2 += red[1][r * 32 + threadIdx.x];
    }
    atomicAdd(&stats[threadIdx.x], t1);
    atomicAdd(&stats[32 + threadIdx.x], t2);
  }
}

__global__ void bn_apply_kernel(const float* __restrict__ z,
                                const float* __restrict__ stats,
                                const float* __restrict__ gamma,
                                const float* __restrict__ beta,
                                float* __restrict__ out) {
  const long long idx = (long long)blockIdx.x * blockDim.x + threadIdx.x;
  if (idx >= (long long)N_ATOMS * 32) return;
  const int c = (int)(idx & 31);
  const float inv_n = 1.f / (float)N_ATOMS;
  const float mu = stats[c] * inv_n;
  const float var = stats[32 + c] * inv_n - mu * mu;
  out[idx] = gamma[c] * (z[idx] - mu) * rsqrtf(var + 1e-5f) + beta[c];
}

__global__ void pool_kernel(const float* __restrict__ h,
                            const int* __restrict__ batch,
                            float* __restrict__ xg) {
  const long long idx = (long long)blockIdx.x * blockDim.x + threadIdx.x;
  if (idx >= (long long)N_ATOMS * 32) return;
  const int node = (int)(idx >> 5), c = (int)(idx & 31);
  atomicAdd(&xg[(size_t)batch[node] * 32 + c], h[idx]);
}

// ---------------- protein branch (vocab-factored conv1d) ----------------
__global__ void conv_table_kernel(const float* __restrict__ emb,
                                  const float* __restrict__ cw,
                                  float* __restrict__ T) {
  const int idx = blockIdx.x * blockDim.x + threadIdx.x;
  if (idx >= VOCAB * KCONV * NF) return;
  const int v = idx / (KCONV * NF);
  const int k = (idx / NF) % KCONV;
  const int f = idx % NF;
  float s = 0.f;
  for (int c = 0; c < EMBD; ++c)
    s += emb[v * EMBD + c] * cw[f * (EMBD * KCONV) + c * KCONV + k];
  T[v * (KCONV * NF) + k * NF + f] = s;
}

__global__ __launch_bounds__(256) void conv_apply_kernel(
    const int* __restrict__ target, const float* __restrict__ T,
    const float* __restrict__ cb, __bf16* __restrict__ cbf) {
  __shared__ float Tl[VOCAB * KCONV * NF];  // 6656 floats = 26 KB
  for (int i = threadIdx.x; i < VOCAB * KCONV * NF; i += 256) Tl[i] = T[i];
  __syncthreads();
  const int gid = blockIdx.x * 256 + threadIdx.x;
  if (gid >= B_G * PCONV) return;
  const int b = gid / PCONV, p = gid % PCONV;
  int tv[KCONV];
#pragma unroll
  for (int k = 0; k < KCONV; ++k) tv[k] = target[b * SEQ + p + k];
#pragma unroll 4
  for (int f = 0; f < NF; ++f) {
    float a = cb[f];
#pragma unroll
    for (int k = 0; k < KCONV; ++k) a += Tl[tv[k] * (KCONV * NF) + k * NF + f];
    cbf[(size_t)b * FLATD + f * PCONV + p] = (__bf16)fmaxf(a, 0.f);
  }
}

// ---------------- conversions / packing ----------------
// src[KxN] f32 row-major -> dst[NxK] bf16 (transposed for B operand)
__global__ void wcvt_t_kernel(const float* __restrict__ src,
                              __bf16* __restrict__ dst, int K, int N) {
  const long long idx = (long long)blockIdx.x * blockDim.x + threadIdx.x;
  if (idx >= (long long)K * N) return;
  const int k = (int)(idx / N), n = (int)(idx % N);
  dst[(size_t)n * K + k] = (__bf16)src[idx];
}

// GIN weight: src[Kx32] f32 -> dst[32xKpad] bf16 transposed, zero-padded
__global__ void gin_wcvt_kernel(const float* __restrict__ src,
                                __bf16* __restrict__ dst, int K, int Kpad) {
  const int idx = blockIdx.x * blockDim.x + threadIdx.x;
  if (idx >= 32 * Kpad) return;
  const int n = idx / Kpad, k = idx % Kpad;
  dst[idx] = (k < K) ? (__bf16)src[k * 32 + n] : (__bf16)0.f;
}

__global__ void pad_x_kernel(const float* __restrict__ x,
                             float* __restrict__ xp) {
  const long long idx = (long long)blockIdx.x * blockDim.x + threadIdx.x;
  if (idx >= (long long)N_ATOMS * F_PAD) return;
  const int n = (int)(idx / F_PAD), c = (int)(idx % F_PAD);
  xp[idx] = (c < F_IN) ? x[(size_t)n * F_IN + c] : 0.f;
}

__global__ void f2bf_kernel(const float* __restrict__ src,
                            __bf16* __restrict__ dst, int n) {
  const int i = blockIdx.x * blockDim.x + threadIdx.x;
  if (i < n) dst[i] = (__bf16)src[i];
}

__global__ void lm_pack_kernel(const float* __restrict__ drug,
                               const float* __restrict__ prot,
                               __bf16* __restrict__ xc) {
  const int idx = blockIdx.x * blockDim.x + threadIdx.x;
  if (idx >= B_G * 2 * LMD) return;
  const int b = idx / (2 * LMD), r = idx % (2 * LMD);
  const float v = (r < LMD) ? drug[b * LMD + r] : prot[b * LMD + (r - LMD)];
  xc[(size_t)b * COMB + 2 * OUTD + r] = (__bf16)v;
}

__global__ void out_gemv_kernel(const float* __restrict__ a2,
                                const float* __restrict__ ow,
                                const float* __restrict__ ob,
                                float* __restrict__ out) {
  const int row = blockIdx.x * 8 + (threadIdx.x >> 5);
  const int lane = threadIdx.x & 31;
  if (row >= B_G) return;
  float s = 0.f;
  for (int c = lane; c < 256; c += 32) s += a2[(size_t)row * 256 + c] * ow[c];
  for (int off = 16; off; off >>= 1) s += __shfl_down(s, off, 32);
  if (lane == 0) out[row] = s + ob[0];
}

// =====================================================================
// host side
// =====================================================================
static inline size_t alignup(size_t x) { return (x + 255) & ~(size_t)255; }

extern "C" void kernel_launch(void* const* d_in, const int* in_sizes, int n_in,
                              void* d_out, int out_size, void* d_ws,
                              size_t ws_size, hipStream_t stream) {
  const float* x       = (const float*)d_in[0];
  const int*   ei      = (const int*)d_in[1];
  const int*   batch   = (const int*)d_in[2];
  const int*   target  = (const int*)d_in[3];
  const float* drug_lm = (const float*)d_in[4];
  const float* prot_lm = (const float*)d_in[5];
  const float* w1a = (const float*)d_in[6];  const float* b1a = (const float*)d_in[7];
  const float* w1b = (const float*)d_in[8];  const float* b1b = (const float*)d_in[9];
  const float* gw_a = (const float*)d_in[10]; const float* gb_a = (const float*)d_in[11];
  const float* gw_b = (const float*)d_in[12]; const float* gb_b = (const float*)d_in[13];
  const float* bn_g = (const float*)d_in[14]; const float* bn_b = (const float*)d_in[15];
  const float* fc1_xd_w = (const float*)d_in[16]; const float* fc1_xd_b = (const float*)d_in[17];
  const float* emb = (const float*)d_in[18];
  const float* convxt_w = (const float*)d_in[19]; const float* convxt_b = (const float*)d_in[20];
  const float* fc1_xt_w = (const float*)d_in[21]; const float* fc1_xt_b = (const float*)d_in[22];
  const float* fc1_w = (const float*)d_in[23]; const float* fc1_b = (const float*)d_in[24];
  const float* fc2_w = (const float*)d_in[25]; const float* fc2_b = (const float*)d_in[26];
  const float* out_w = (const float*)d_in[27]; const float* out_b = (const float*)d_in[28];

  char* ws = (char*)d_ws;
  size_t off = 0;
  auto take = [&](size_t bytes) { char* p = ws + off; off += alignup(bytes); return p; };
  float*  agg   = (float*)take((size_t)N_ATOMS * F_PAD * 4);
  float*  x96   = (float*)take((size_t)N_ATOMS * F_PAD * 4);
  float*  h_a   = (float*)take((size_t)N_ATOMS * DIM * 4);
  float*  h_b   = (float*)take((size_t)N_ATOMS * DIM * 4);
  float*  ztmp  = (float*)take((size_t)N_ATOMS * DIM * 4);
  float*  stats = (float*)take(64 * 4);
  float*  xg    = (float*)take((size_t)B_G * DIM * 4);
  __bf16* xgb   = (__bf16*)take((size_t)B_G * DIM * 2);
  float*  Ttab  = (float*)take((size_t)VOCAB * KCONV * NF * 4);
  __bf16* cbf   = (__bf16*)take((size_t)B_G * FLATD * 2);
  __bf16* xc    = (__bf16*)take((size_t)B_G * COMB * 2);
  __bf16* a1    = (__bf16*)take((size_t)B_G * 1024 * 2);
  float*  a2    = (float*)take((size_t)B_G * 256 * 4);
  __bf16* wxt_t = (__bf16*)take((size_t)FLATD * OUTD * 2);   // [128 x 31776]
  __bf16* wf1_t = (__bf16*)take((size_t)COMB * 1024 * 2);    // [1024 x 2304]
  __bf16* wf2_t = (__bf16*)take((size_t)1024 * 256 * 2);     // [256 x 1024]
  __bf16* wxd_t = (__bf16*)take((size_t)DIM * OUTD * 2);     // [128 x 32]
  __bf16* w1a_t = (__bf16*)take((size_t)32 * F_PAD * 2);     // padded [32 x 96]
  __bf16* w1b_t = (__bf16*)take((size_t)32 * 32 * 2);
  __bf16* gwa_t = (__bf16*)take((size_t)4 * 32 * 32 * 2);
  __bf16* gwb_t = (__bf16*)take((size_t)4 * 32 * 32 * 2);

  auto cvt_t = [&](const float* s, __bf16* d, int K, int N) {
    const long long n = (long long)K * N;
    wcvt_t_kernel<<<(int)((n + 255) / 256), 256, 0, stream>>>(s, d, K, N);
  };

  // -- weight conversion + transpose (one-time, L2-resident thereafter) --
  cvt_t(fc1_xt_w, wxt_t, FLATD, OUTD);
  cvt_t(fc1_w, wf1_t, COMB, 1024);
  cvt_t(fc2_w, wf2_t, 1024, 256);
  cvt_t(fc1_xd_w, wxd_t, DIM, OUTD);
  gin_wcvt_kernel<<<(32 * F_PAD + 255) / 256, 256, 0, stream>>>(w1a, w1a_t,
                                                                F_IN, F_PAD);
  gin_wcvt_kernel<<<4, 256, 0, stream>>>(w1b, w1b_t, 32, 32);
  for (int i = 0; i < 4; ++i) {
    gin_wcvt_kernel<<<4, 256, 0, stream>>>(gw_a + i * 1024, gwa_t + i * 1024, 32, 32);
    gin_wcvt_kernel<<<4, 256, 0, stream>>>(gw_b + i * 1024, gwb_t + i * 1024, 32, 32);
  }

  // -- protein branch: vocab-factored conv1d --
  conv_table_kernel<<<(VOCAB * KCONV * NF + 255) / 256, 256, 0, stream>>>(
      emb, convxt_w, Ttab);
  conv_apply_kernel<<<(B_G * PCONV + 255) / 256, 256, 0, stream>>>(
      target, Ttab, convxt_b, cbf);

  // -- GIN stack --
  const int ntiles = N_ATOMS / 16;           // 6250
  const int gin_blocks = (ntiles + 7) / 8;

  // layer 1 (78 -> 32), padded to K=96
  pad_x_kernel<<<(int)(((long long)N_ATOMS * F_PAD + 255) / 256), 256, 0,
                 stream>>>(x, x96);
  hipMemsetAsync(agg, 0, (size_t)N_ATOMS * F_PAD * 4, stream);
  hipMemsetAsync(stats, 0, 64 * 4, stream);
  {
    const long long tot = (long long)E_EDGES * F_IN;
    scatter_kernel<<<(int)((tot + 255) / 256), 256, 0, stream>>>(
        x, ei, agg, F_IN, F_IN, F_PAD);
  }
  gin_mlp_wmma<<<gin_blocks, 256, 0, stream>>>(x96, agg, F_PAD, w1a_t, b1a,
                                               w1b_t, b1b, ztmp, ntiles);
  bn_stats_kernel<<<256, 256, 0, stream>>>(ztmp, stats);
  bn_apply_kernel<<<(N_ATOMS * 32 + 255) / 256, 256, 0, stream>>>(
      ztmp, stats, bn_g, bn_b, h_a);

  // layers 2..5 (32 -> 32)
  float* hc = h_a;
  float* hn = h_b;
  for (int li = 1; li <= 4; ++li) {
    hipMemsetAsync(agg, 0, (size_t)N_ATOMS * DIM * 4, stream);
    hipMemsetAsync(stats, 0, 64 * 4, stream);
    const long long tot = (long long)E_EDGES * DIM;
    scatter_kernel<<<(int)((tot + 255) / 256), 256, 0, stream>>>(
        hc, ei, agg, DIM, DIM, DIM);
    gin_mlp_wmma<<<gin_blocks, 256, 0, stream>>>(
        hc, agg, DIM, gwa_t + (li - 1) * 1024, gb_a + (li - 1) * DIM,
        gwb_t + (li - 1) * 1024, gb_b + (li - 1) * DIM, ztmp, ntiles);
    bn_stats_kernel<<<256, 256, 0, stream>>>(ztmp, stats);
    bn_apply_kernel<<<(N_ATOMS * 32 + 255) / 256, 256, 0, stream>>>(
        ztmp, stats, bn_g + li * DIM, bn_b + li * DIM, hn);
    float* t = hc; hc = hn; hn = t;
  }

  // -- global add pool + fc1_xd -> xc[:, 0:128] --
  hipMemsetAsync(xg, 0, (size_t)B_G * DIM * 4, stream);
  pool_kernel<<<(N_ATOMS * 32 + 255) / 256, 256, 0, stream>>>(hc, batch, xg);
  f2bf_kernel<<<(B_G * DIM + 255) / 256, 256, 0, stream>>>(xg, xgb, B_G * DIM);

  // NT=2: 16x32 tile per wave (N=128 -> 4 groups, 256 waves)
  {
    const int tiles = (B_G / 16) * (OUTD / 32);
    gemm_bf16_wmma<2, true><<<(tiles + 7) / 8, 256, 0, stream>>>(
        xgb, wxd_t, fc1_xd_b, xc, B_G, OUTD, DIM, DIM, COMB, 0);
  }

  // -- fc1_xt -> xc[:, 128:256]  (K = 31776 = 993 exact WMMA steps) --
  {
    const int tiles = (B_G / 16) * (OUTD / 32);
    gemm_bf16_wmma<2, true><<<(tiles + 7) / 8, 256, 0, stream>>>(
        cbf, wxt_t, fc1_xt_b, xc, B_G, OUTD, FLATD, FLATD, COMB, OUTD);
  }

  // -- LM embeddings -> xc[:, 256:2304] --
  lm_pack_kernel<<<(B_G * 2 * LMD + 255) / 256, 256, 0, stream>>>(drug_lm,
                                                                  prot_lm, xc);

  // -- head: fc1 (2304->1024, NT=4) -> fc2 (1024->256, NT=2) -> out --
  {
    const int tiles = (B_G / 16) * (1024 / 64);
    gemm_bf16_wmma<4, true><<<(tiles + 7) / 8, 256, 0, stream>>>(
        xc, wf1_t, fc1_b, a1, B_G, 1024, COMB, COMB, 1024, 0);
  }
  {
    const int tiles = (B_G / 16) * (256 / 32);
    gemm_bf16_wmma<2, false><<<(tiles + 7) / 8, 256, 0, stream>>>(
        a1, wf2_t, fc2_b, a2, B_G, 256, 1024, 1024, 256, 0);
  }
  out_gemv_kernel<<<B_G / 8, 256, 0, stream>>>(a2, out_w, out_b,
                                               (float*)d_out);
}
